// FlaxXerxesDecoderLayer_55748675502250
// MI455X (gfx1250) — matrix operand order repaired and checked by reading, not compiled, not run on hardware
//
#include <hip/hip_runtime.h>
#include <hip/hip_bf16.h>
#include <cstdint>
#include <cstddef>

// ---------------------------------------------------------------------------
// Problem sizes (FlaxXerxesDecoderLayer reference)
// ---------------------------------------------------------------------------
constexpr int S   = 4096;
constexpr int H   = 2048;
constexpr int NH  = 8;
constexpr int NKV = 4;
constexpr int HD  = 256;
constexpr int I   = 8192;
constexpr int WINDOW = 4096;
constexpr float EPS = 1e-6f;

typedef __bf16 v16bf __attribute__((ext_vector_type(16)));
typedef __bf16 v8bf  __attribute__((ext_vector_type(8)));
typedef float  v8f   __attribute__((ext_vector_type(8)));
typedef unsigned v4u __attribute__((ext_vector_type(4)));
typedef int    v8i   __attribute__((ext_vector_type(8)));
typedef int    v4i   __attribute__((ext_vector_type(4)));

#define AS1 __attribute__((address_space(1)))
#define AS3 __attribute__((address_space(3)))

#ifndef __has_builtin
#define __has_builtin(x) 0
#endif

#if __has_builtin(__builtin_amdgcn_global_load_async_to_lds_b128)
#define USE_ASYNC_LDS 1
#else
#define USE_ASYNC_LDS 0
#endif

#if __has_builtin(__builtin_amdgcn_tensor_load_to_lds) && \
    __has_builtin(__builtin_amdgcn_s_wait_tensorcnt)
#define USE_TDM 1
#else
#define USE_TDM 0
#endif

static __device__ __forceinline__ __bf16 f2bf(float f) {
    return (__bf16)f;   // native v_cvt path on gfx1250 (RNE)
}

// 16B global -> LDS copy, async-to-LDS when available
static __device__ __forceinline__ void cp16_g2l(const void* g, void* l) {
#if USE_ASYNC_LDS
    __builtin_amdgcn_global_load_async_to_lds_b128((AS1 v4i*)g, (AS3 v4i*)l, 0, 0);
#else
    *(uint4*)l = *(const uint4*)g;
#endif
}

static __device__ __forceinline__ void wait_async0() {
#if USE_ASYNC_LDS
#if __has_builtin(__builtin_amdgcn_s_wait_asynccnt)
    __builtin_amdgcn_s_wait_asynccnt(0);
#else
    asm volatile("s_wait_asynccnt 0x0" ::: "memory");
#endif
#endif
}

static __device__ __forceinline__ v16bf join16(v8bf lo, v8bf hi) {
    return __builtin_shufflevector(lo, hi, 0,1,2,3,4,5,6,7,8,9,10,11,12,13,14,15);
}

static __device__ __forceinline__ v8f wmma_bf16(v16bf a, v16bf b, v8f c) {
    // D = A(16x32 bf16) * B(32x16 bf16) + C(16x16 f32)
    return __builtin_amdgcn_wmma_f32_16x16x32_bf16(false, a, false, b, (short)0, c,
                                                   false, false);
}

#if USE_TDM
// TDM: DMA a rows x 256 bf16 tile (row-major, row stride 256) from global to LDS.
static __device__ __forceinline__ void tdm_load_tile_256bf16(const __bf16* gsrc,
                                                             __bf16* lds_dst,
                                                             int rows) {
    const unsigned lds_off = (unsigned)(size_t)(AS3 __bf16*)lds_dst;
    const unsigned long long ga = (unsigned long long)(size_t)gsrc;
    v4u g0;
    g0[0] = 1u;                                         // count=1, user descriptor
    g0[1] = lds_off;                                    // lds_addr
    g0[2] = (unsigned)(ga & 0xFFFFFFFFu);               // global_addr[31:0]
    g0[3] = (unsigned)((ga >> 32) & 0x01FFFFFFu)        // global_addr[56:32]
            | (2u << 30);                               // type = 2 ("image")
    v8i g1;
    g1[0] = (int)(1u << 16);                            // data_size = 2 bytes
    g1[1] = (int)(256u << 16);                          // tensor_dim0 = 256 (lo16)
    g1[2] = (int)((unsigned)(S & 0xFFFF) << 16);        // tensor_dim1 lo16
    g1[3] = (int)(256u << 16);                          // tile_dim0 = 256
    g1[4] = rows & 0xFFFF;                              // tile_dim1 = rows
    g1[5] = 256;                                        // tensor_dim0_stride = 256
    g1[6] = 0;
    g1[7] = 0;
    v4i zz4 = {0, 0, 0, 0};
    v8i zz8 = {0, 0, 0, 0, 0, 0, 0, 0};
    __builtin_amdgcn_tensor_load_to_lds(g0, g1, zz4, zz4, zz8, 0);
}
#endif

// ---------------------------------------------------------------------------
// RMSNorm: fp32 row -> bf16 normalized row (one block per row, 256 threads)
// ---------------------------------------------------------------------------
__global__ __launch_bounds__(256) void rmsnorm_kernel(const float* __restrict__ x,
                                                      const float* __restrict__ w,
                                                      __bf16* __restrict__ out,
                                                      int ncols) {
    const int row = blockIdx.x;
    const int tid = threadIdx.x;
    const float* xr = x + (size_t)row * ncols;

    float ss = 0.f;
    for (int c = tid; c < ncols; c += 256) { float v = xr[c]; ss += v * v; }

    __shared__ float red[256];
    red[tid] = ss;
    __syncthreads();
    for (int s = 128; s > 0; s >>= 1) {
        if (tid < s) red[tid] += red[tid + s];
        __syncthreads();
    }
    const float inv = rsqrtf(red[0] / (float)ncols + EPS);
    for (int c = tid; c < ncols; c += 256)
        out[(size_t)row * ncols + c] = f2bf(xr[c] * inv * w[c]);
}

// ---------------------------------------------------------------------------
// WMMA GEMM: C[M,N] (f32) = A[M,K] (bf16, row-major) x B[K,N] (f32 -> bf16)
//            optionally + residual[M,N]
// Block tile: 64 x 128, BK = 32.  256 threads = 8 waves.
// ---------------------------------------------------------------------------
#define GBM 64
#define GBN 128
#define GBK 32
#define ASTR 40
#define BSTR 40

__global__ __launch_bounds__(256) void gemm_bf16_wmma(const __bf16* __restrict__ A,
                                                      const float*  __restrict__ Bw,
                                                      const float*  __restrict__ resid,
                                                      float* __restrict__ C,
                                                      int M, int N, int K) {
    __shared__ __attribute__((aligned(16))) __bf16 As[GBM * ASTR];
    __shared__ __attribute__((aligned(16))) __bf16 Bt[GBN * BSTR]; // [n][k]

    const int tid  = threadIdx.x;
    const int lane = tid & 31;
    const int wv   = tid >> 5;
    const int m0   = blockIdx.y * GBM;
    const int n0   = blockIdx.x * GBN;

    const int rt = wv >> 1;          // row tile 0..3
    const int cg = wv & 1;           // col group 0..1 (4 tiles each)

    v8f acc[4] = {};

    for (int k0 = 0; k0 < K; k0 += GBK) {
        // ---- stage A tile: 64x32 bf16 (raw copy, async path when available)
        {
            const int idx = tid * 8;
            const int r = idx >> 5, c = idx & 31;
            cp16_g2l(&A[(size_t)(m0 + r) * K + k0 + c], &As[r * ASTR + c]);
        }
        // ---- stage B tile transposed: 32x128 f32 -> Bt[n][k] bf16 (VALU convert)
        #pragma unroll
        for (int i = 0; i < 4; ++i) {
            const int idx = (i * 256 + tid) * 4;
            const int k = idx >> 7, n = idx & 127;
            const float4 v = *(const float4*)&Bw[(size_t)(k0 + k) * N + n0 + n];
            Bt[(n + 0) * BSTR + k] = f2bf(v.x);
            Bt[(n + 1) * BSTR + k] = f2bf(v.y);
            Bt[(n + 2) * BSTR + k] = f2bf(v.z);
            Bt[(n + 3) * BSTR + k] = f2bf(v.w);
        }
        if (k0 + GBK < K) {
            __builtin_prefetch(&A[(size_t)(m0 + (tid >> 2)) * K + k0 + GBK], 0, 1);
            __builtin_prefetch(&Bw[(size_t)(k0 + GBK + (tid >> 5)) * N + n0 + (tid & 31) * 4], 0, 1);
        }
        wait_async0();
        __syncthreads();

        // ---- A fragment (16x32 bf16): lanes<16 K0..7/16..23, else 8..15/24..31
        const int ar = rt * 16 + (lane & 15);
        const int kh = (lane < 16) ? 0 : 8;
        const v8bf alo = *(const v8bf*)&As[ar * ASTR + kh];
        const v8bf ahi = *(const v8bf*)&As[ar * ASTR + kh + 16];
        const v16bf af = join16(alo, ahi);

        #pragma unroll
        for (int j = 0; j < 4; ++j) {
            const int bc = (cg * 4 + j) * 16 + (lane & 15);
            const int ks = (lane < 16) ? 0 : 8;
            const v8bf blo = *(const v8bf*)&Bt[bc * BSTR + ks];
            const v8bf bhi = *(const v8bf*)&Bt[bc * BSTR + ks + 16];
            acc[j] = wmma_bf16(af, join16(blo, bhi), acc[j]);
        }
        __syncthreads();
    }

    // ---- epilogue
    const bool hasres = (resid != nullptr);
    #pragma unroll
    for (int j = 0; j < 4; ++j) {
        const int ctile = (cg * 4 + j) * 16;
        #pragma unroll
        for (int i = 0; i < 8; ++i) {
            const int row = m0 + rt * 16 + ((lane >= 16) ? 8 : 0) + i;
            const int col = n0 + ctile + (lane & 15);
            const size_t idx = (size_t)row * N + col;
            C[idx] = acc[j][i] + (hasres ? resid[idx] : 0.f);
        }
    }
}

// ---------------------------------------------------------------------------
// RoPE + head-major bf16 packing.
// ---------------------------------------------------------------------------
__global__ __launch_bounds__(256) void rope_pack_kernel(const float* __restrict__ qf,
                                                        const float* __restrict__ kf,
                                                        const float* __restrict__ vf,
                                                        const int*   __restrict__ pos_ids,
                                                        __bf16* __restrict__ qb,
                                                        __bf16* __restrict__ kb,
                                                        __bf16* __restrict__ vb) {
    const int s   = blockIdx.x;
    const int tid = threadIdx.x;
    const float pos = (float)pos_ids[s];
    const float kfreq = -2.f * 9.210340372f / (float)HD;  // -2*ln(10000)/HD

    #pragma unroll
    for (int i = 0; i < 4; ++i) {               // Q: 1024 rotation pairs
        const int idx = tid + i * 256;
        const int h = idx >> 7, d = idx & 127;
        const float fr = pos * __expf((float)d * kfreq);
        const float c = __cosf(fr), sn = __sinf(fr);
        const float x1 = qf[(size_t)s * (NH * HD) + h * HD + d];
        const float x2 = qf[(size_t)s * (NH * HD) + h * HD + d + 128];
        __bf16* dst = qb + (size_t)h * S * HD + (size_t)s * HD;
        dst[d]       = f2bf(x1 * c - x2 * sn);
        dst[d + 128] = f2bf(x2 * c + x1 * sn);
    }
    #pragma unroll
    for (int i = 0; i < 2; ++i) {               // K: 512 pairs
        const int idx = tid + i * 256;
        const int h = idx >> 7, d = idx & 127;
        const float fr = pos * __expf((float)d * kfreq);
        const float c = __cosf(fr), sn = __sinf(fr);
        const float x1 = kf[(size_t)s * (NKV * HD) + h * HD + d];
        const float x2 = kf[(size_t)s * (NKV * HD) + h * HD + d + 128];
        __bf16* dst = kb + (size_t)h * S * HD + (size_t)s * HD;
        dst[d]       = f2bf(x1 * c - x2 * sn);
        dst[d + 128] = f2bf(x2 * c + x1 * sn);
    }
    #pragma unroll
    for (int i = 0; i < 4; ++i) {               // V: convert 1024 elems
        const int idx = tid + i * 256;
        const int h = idx >> 8, d = idx & 255;
        vb[(size_t)h * S * HD + (size_t)s * HD + d] =
            f2bf(vf[(size_t)s * (NKV * HD) + h * HD + d]);
    }
}

// ---------------------------------------------------------------------------
// Flash attention (WMMA bf16) — one block per (head, 32-row q tile).
// ---------------------------------------------------------------------------
#define QT 32
#define KT 32
#define QSTR 264
#define VSTR 40
#define PSTR 40
#define SSTR 33

__global__ __launch_bounds__(256) void attn_wmma_kernel(const __bf16* __restrict__ qb,
                                                        const __bf16* __restrict__ kb,
                                                        const __bf16* __restrict__ vb,
                                                        const int*    __restrict__ amask,
                                                        __bf16* __restrict__ attnb) {
    __shared__ __attribute__((aligned(16))) __bf16 Qs[QT * QSTR];
    __shared__ __attribute__((aligned(16))) __bf16 Ks[KT * 256];   // TDM tile, unpadded
    __shared__ __attribute__((aligned(16))) __bf16 Vt[HD * VSTR];  // [d][k]
    __shared__ __attribute__((aligned(16))) __bf16 Ps[QT * PSTR];
    __shared__ float Ss[QT * SSTR];
    __shared__ float mrow[QT], lrow[QT], arow[QT];

    const int h  = blockIdx.x;
    const int q0 = blockIdx.y * QT;
    const int kvh = h >> 1;                       // GQA groups = 2
    const __bf16* Q = qb + (size_t)h   * S * HD;
    const __bf16* K = kb + (size_t)kvh * S * HD;
    const __bf16* V = vb + (size_t)kvh * S * HD;

    const int tid = threadIdx.x, lane = tid & 31, wv = tid >> 5;
    const int rw = wv & 1;
    const int cg = wv >> 1;

    // Q tile: 32x256 bf16, async 16B chunks
    #pragma unroll
    for (int i = 0; i < 4; ++i) {
        const int idx = (tid + i * 256) * 8;
        const int r = idx >> 8, d = idx & 255;
        cp16_g2l(&Q[(size_t)(q0 + r) * HD + d], &Qs[r * QSTR + d]);
    }
    if (tid < QT) { mrow[tid] = -INFINITY; lrow[tid] = 0.f; }
    wait_async0();

    v8f acc[4] = {};
    const float scale = 0.0625f;                  // HD^-0.5

    int kstart = q0 + QT - WINDOW;
    if (kstart < 0) kstart = 0;
    kstart &= ~(KT - 1);
    const int kend = q0 + QT;

    for (int kt = kstart; kt < kend; kt += KT) {
        __syncthreads();
        // ---- stage K tile (TDM when available, else async/manual copies)
#if USE_TDM
        if (tid < 32) {
            tdm_load_tile_256bf16(&K[(size_t)kt * HD], &Ks[0], KT);
        }
#else
        #pragma unroll
        for (int i = 0; i < 4; ++i) {
            const int idx = (tid + i * 256) * 8;
            const int r = idx >> 8, d = idx & 255;
            cp16_g2l(&K[(size_t)(kt + r) * HD + d], &Ks[r * 256 + d]);
        }
#endif
        // ---- stage V tile transposed (needs data reshuffle -> VALU path)
        #pragma unroll
        for (int i = 0; i < 32; ++i) {
            const int idx = tid + i * 256;
            const int r = idx >> 8, d = idx & 255;
            Vt[d * VSTR + r] = V[(size_t)(kt + r) * HD + d];
        }
#if USE_TDM
        if (tid < 32) { __builtin_amdgcn_s_wait_tensorcnt(0); }
#else
        wait_async0();
#endif
        __syncthreads();

        // ---- S = Q K^T * scale : 2x2 tiles of 16x16, waves 0..3
        if (wv < 4) {
            const int sr = (wv & 1) * 16, sc = (wv >> 1) * 16;
            v8f sacc = {};
            #pragma unroll
            for (int d0 = 0; d0 < HD; d0 += 32) {
                const int ar = sr + (lane & 15);
                const int kh = (lane < 16) ? 0 : 8;
                const v8bf alo = *(const v8bf*)&Qs[ar * QSTR + d0 + kh];
                const v8bf ahi = *(const v8bf*)&Qs[ar * QSTR + d0 + kh + 16];
                const int br = sc + (lane & 15);
                const int ks = (lane < 16) ? 0 : 8;
                const v8bf blo = *(const v8bf*)&Ks[br * 256 + d0 + ks];
                const v8bf bhi = *(const v8bf*)&Ks[br * 256 + d0 + ks + 16];
                sacc = wmma_bf16(join16(alo, ahi), join16(blo, bhi), sacc);
            }
            #pragma unroll
            for (int i = 0; i < 8; ++i) {
                const int r = sr + ((lane >= 16) ? 8 : 0) + i;
                const int c = sc + (lane & 15);
                Ss[r * SSTR + c] = sacc[i] * scale;
            }
        }
        __syncthreads();

        // ---- online softmax (one thread per q row)
        if (tid < QT) {
            const int r = tid, grow = q0 + r;
            const float mold = mrow[r];
            float tmax = -INFINITY;
            float sv[KT];
            #pragma unroll
            for (int c = 0; c < KT; ++c) {
                const int gc = kt + c;
                const bool keep = (gc <= grow) && (gc > grow - WINDOW) && (amask[gc] > 0);
                const float sc = keep ? Ss[r * SSTR + c] : -INFINITY;
                sv[c] = sc;
                tmax = fmaxf(tmax, sc);
            }
            const float mn = fmaxf(mold, tmax);
            const float al = __expf(mold - mn);
            float psum = 0.f;
            #pragma unroll
            for (int c = 0; c < KT; ++c) {
                const float p = (sv[c] == -INFINITY) ? 0.f : __expf(sv[c] - mn);
                Ps[r * PSTR + c] = f2bf(p);
                psum += p;
            }
            lrow[r] = lrow[r] * al + psum;
            mrow[r] = mn;
            arow[r] = al;
        }
        __syncthreads();

        // ---- O = O*alpha + P V  (one 16x16x32 WMMA per output tile)
        {
            const int ar = rw * 16 + (lane & 15);
            const int kh = (lane < 16) ? 0 : 8;
            const v8bf alo = *(const v8bf*)&Ps[ar * PSTR + kh];
            const v8bf ahi = *(const v8bf*)&Ps[ar * PSTR + kh + 16];
            const v16bf pf = join16(alo, ahi);
            #pragma unroll
            for (int j = 0; j < 4; ++j) {
                #pragma unroll
                for (int i = 0; i < 8; ++i) {
                    const int r = rw * 16 + ((lane >= 16) ? 8 : 0) + i;
                    acc[j][i] *= arow[r];
                }
                const int bc = (cg * 4 + j) * 16 + (lane & 15);
                const int ks = (lane < 16) ? 0 : 8;
                const v8bf blo = *(const v8bf*)&Vt[bc * VSTR + ks];
                const v8bf bhi = *(const v8bf*)&Vt[bc * VSTR + ks + 16];
                acc[j] = wmma_bf16(pf, join16(blo, bhi), acc[j]);
            }
        }
    }
    __syncthreads();

    // ---- finalize
    #pragma unroll
    for (int j = 0; j < 4; ++j) {
        const int ctile = (cg * 4 + j) * 16;
        #pragma unroll
        for (int i = 0; i < 8; ++i) {
            const int r = rw * 16 + ((lane >= 16) ? 8 : 0) + i;
            const int c = ctile + (lane & 15);
            attnb[(size_t)(q0 + r) * (NH * HD) + h * HD + c] = f2bf(acc[j][i] / lrow[r]);
        }
    }
}

// ---------------------------------------------------------------------------
// gated-GeLU (tanh approx):  hb = bf16( gelu(g) * u )
// ---------------------------------------------------------------------------
__global__ __launch_bounds__(256) void gelu_mul_kernel(const float* __restrict__ g,
                                                       const float* __restrict__ u,
                                                       __bf16* __restrict__ o,
                                                       int n) {
    const int i = blockIdx.x * 256 + threadIdx.x;
    if (i < n) {
        const float x = g[i];
        const float t = tanhf(0.7978845608f * (x + 0.044715f * x * x * x));
        o[i] = f2bf(0.5f * x * (1.f + t) * u[i]);
    }
}

// ---------------------------------------------------------------------------
// Host launcher
// ---------------------------------------------------------------------------
extern "C" void kernel_launch(void* const* d_in, const int* in_sizes, int n_in,
                              void* d_out, int out_size, void* d_ws, size_t ws_size,
                              hipStream_t stream) {
    (void)in_sizes; (void)n_in; (void)out_size; (void)ws_size;

    const float* hidden  = (const float*)d_in[0];
    const float* q_w     = (const float*)d_in[1];
    const float* k_w     = (const float*)d_in[2];
    const float* v_w     = (const float*)d_in[3];
    const float* o_w     = (const float*)d_in[4];
    const float* gate_w  = (const float*)d_in[5];
    const float* up_w    = (const float*)d_in[6];
    const float* down_w  = (const float*)d_in[7];
    const float* ln1_w   = (const float*)d_in[8];
    const float* ln2_w   = (const float*)d_in[9];
    const int*   amask   = (const int*)d_in[10];
    const int*   pos_ids = (const int*)d_in[11];
    float* out = (float*)d_out;

    char* ws = (char*)d_ws;
    size_t off = 0;
    auto alloc = [&](size_t bytes) { void* p = ws + off; off += (bytes + 255) & ~(size_t)255; return p; };
    __bf16* xn      = (__bf16*)alloc((size_t)S * H * 2);
    __bf16* xn2     = (__bf16*)alloc((size_t)S * H * 2);
    float*  qf      = (float*) alloc((size_t)S * NH  * HD * 4);
    float*  kf      = (float*) alloc((size_t)S * NKV * HD * 4);
    float*  vf      = (float*) alloc((size_t)S * NKV * HD * 4);
    __bf16* qb      = (__bf16*)alloc((size_t)S * NH  * HD * 2);
    __bf16* kb      = (__bf16*)alloc((size_t)S * NKV * HD * 2);
    __bf16* vb      = (__bf16*)alloc((size_t)S * NKV * HD * 2);
    __bf16* attnb   = (__bf16*)alloc((size_t)S * H * 2);
    float*  hidden2 = (float*) alloc((size_t)S * H * 4);
    float*  gf      = (float*) alloc((size_t)S * I * 4);
    float*  uf      = (float*) alloc((size_t)S * I * 4);
    __bf16* hb      = (__bf16*)alloc((size_t)S * I * 2);

    rmsnorm_kernel<<<S, 256, 0, stream>>>(hidden, ln1_w, xn, H);

    gemm_bf16_wmma<<<dim3((NH * HD) / GBN, S / GBM), 256, 0, stream>>>(xn, q_w, nullptr, qf, S, NH * HD, H);
    gemm_bf16_wmma<<<dim3((NKV * HD) / GBN, S / GBM), 256, 0, stream>>>(xn, k_w, nullptr, kf, S, NKV * HD, H);
    gemm_bf16_wmma<<<dim3((NKV * HD) / GBN, S / GBM), 256, 0, stream>>>(xn, v_w, nullptr, vf, S, NKV * HD, H);

    rope_pack_kernel<<<S, 256, 0, stream>>>(qf, kf, vf, pos_ids, qb, kb, vb);

    attn_wmma_kernel<<<dim3(NH, S / QT), 256, 0, stream>>>(qb, kb, vb, amask, attnb);

    gemm_bf16_wmma<<<dim3(H / GBN, S / GBM), 256, 0, stream>>>(attnb, o_w, hidden, hidden2, S, H, H);

    rmsnorm_kernel<<<S, 256, 0, stream>>>(hidden2, ln2_w, xn2, H);

    gemm_bf16_wmma<<<dim3(I / GBN, S / GBM), 256, 0, stream>>>(xn2, gate_w, nullptr, gf, S, I, H);
    gemm_bf16_wmma<<<dim3(I / GBN, S / GBM), 256, 0, stream>>>(xn2, up_w,   nullptr, uf, S, I, H);

    gelu_mul_kernel<<<(S * I + 255) / 256, 256, 0, stream>>>(gf, uf, hb, S * I);

    gemm_bf16_wmma<<<dim3(H / GBN, S / GBM), 256, 0, stream>>>(hb, down_w, hidden2, out, S, H, I);
}
